// FastMHA_80539226735288
// MI455X (gfx1250) — matrix operand order repaired and checked by reading, not compiled, not run on hardware
//
#include <hip/hip_runtime.h>
#include <stdint.h>

// Problem constants (from reference): B=2, S=2048, D=1024, H=16, DH=64.
#define S_LEN 2048
#define D_DIM 1024
#define DH_DIM 64
#define BH 32            // B*H heads; each head = contiguous [S][DH] block (raw-reshape quirk)
#define MROWS 4096       // B*S flattened rows for the projections
#define NEGV (-1.0e9f)
#define SM_SCALE 0.125f  // 1/sqrt(DH)

typedef __attribute__((ext_vector_type(16))) __bf16 v16bf;
typedef __attribute__((ext_vector_type(8)))  __bf16 v8bf;
typedef __attribute__((ext_vector_type(8)))  float  v8f;

static __device__ __forceinline__ v16bf cat8(v8bf lo, v8bf hi) {
  v16bf r;
#pragma unroll
  for (int i = 0; i < 8; ++i) { r[i] = lo[i]; r[i + 8] = hi[i]; }
  return r;
}

static __device__ __forceinline__ v8f vzero8() {
  v8f z;
#pragma unroll
  for (int e = 0; e < 8; ++e) z[e] = 0.f;
  return z;
}

static __device__ __forceinline__ v8f wmma_bf16(v16bf a, v16bf b, v8f c) {
  // D = A(16x32,bf16) * B(32x16,bf16) + C(16x16,f32)
  return __builtin_amdgcn_wmma_f32_16x16x32_bf16(false, a, false, b, (short)0, c,
                                                 false, false);
}

// A fragment: 16(M)x32(K) from a row-major MxK source (row stride in elements).
static __device__ __forceinline__ v16bf load_a(const __bf16* base, int stride, int lane) {
  const int row = lane & 15, kh = (lane >> 4) << 3;
  const __bf16* p = base + row * stride + kh;
  return cat8(*(const v8bf*)p, *(const v8bf*)(p + 16));
}

// B fragment: 32(K)x16(N) from a row-major NxK source (i.e. B^T; stride in elements).
static __device__ __forceinline__ v16bf load_b(const __bf16* base, int stride, int lane) {
  const int n = lane & 15, kh = (lane >> 4) << 3;
  const __bf16* p = base + n * stride + kh;
  return cat8(*(const v8bf*)p, *(const v8bf*)(p + 16));
}

// Async DMA: copy 32 contiguous bytes global -> LDS (2x b128, ASYNCcnt-tracked).
// ISA 08_async_tensor: the LDS address is the VDST VGPR plus the instruction offset, and the same
// INST_OFFSET is added to the global address, so one reg pair + offset:16 works.
// The LDS byte offset is the low 32 bits of the generic pointer (aperture rule).
static __device__ __forceinline__ void async_copy32(const __bf16* g, __bf16* l) {
  unsigned lo = (unsigned)(size_t)l;
  uint64_t ga = (uint64_t)(size_t)g;
  asm volatile("global_load_async_to_lds_b128 %0, %1, off\n\t"
               "global_load_async_to_lds_b128 %0, %1, off offset:16"
               :: "v"(lo), "v"(ga) : "memory");
}
static __device__ __forceinline__ void wait_async0() {
  asm volatile("s_wait_asynccnt 0" ::: "memory");
}

// ---------------------------------------------------------------------------
// Kernel 1: out[n,m] = sum_k X[n,k]*W[m,k] + b[m], written as bf16.
// Block tile 128x128, 8 waves (2M x 4N), K-step 32, double-buffered LDS so the
// next tile's global loads overlap the current tile's WMMAs (1 barrier/iter).
// ---------------------------------------------------------------------------
__global__ __launch_bounds__(256)
void proj_gemm(const float* __restrict__ X, const float* __restrict__ W,
               const float* __restrict__ bias, __bf16* __restrict__ out) {
  __shared__ __align__(16) __bf16 xa[2][128 * 32];
  __shared__ __align__(16) __bf16 wb[2][128 * 32];

  const int tid  = threadIdx.x;
  const int lane = tid & 31, wave = tid >> 5;
  const int wtm = wave & 1, wtn = wave >> 1;            // wave -> (Mhalf, Nquarter)
  const int BM = blockIdx.y * 128, BN = blockIdx.x * 128;

  v8f acc[4][2];
#pragma unroll
  for (int ms = 0; ms < 4; ++ms)
#pragma unroll
    for (int ns = 0; ns < 2; ++ns) acc[ms][ns] = vzero8();

  const int srow = tid >> 1, shalf = (tid & 1) << 4;    // staging map: 16 elems/thread/tile
  const float* xs = X + (size_t)(BM + srow) * D_DIM + shalf;
  const float* ws = W + (size_t)(BN + srow) * D_DIM + shalf;

  auto stage = [&](int kk, int buf) {
    float tx[16], tw[16];
#pragma unroll
    for (int q = 0; q < 4; ++q) {
      *(float4*)(tx + 4 * q) = *(const float4*)(xs + kk + 4 * q);
      *(float4*)(tw + 4 * q) = *(const float4*)(ws + kk + 4 * q);
    }
    v8bf bx0, bx1, bw0, bw1;
#pragma unroll
    for (int e = 0; e < 8; ++e) {
      bx0[e] = (__bf16)tx[e]; bx1[e] = (__bf16)tx[e + 8];
      bw0[e] = (__bf16)tw[e]; bw1[e] = (__bf16)tw[e + 8];
    }
    *(v8bf*)(&xa[buf][srow * 32 + shalf])     = bx0;
    *(v8bf*)(&xa[buf][srow * 32 + shalf + 8]) = bx1;
    *(v8bf*)(&wb[buf][srow * 32 + shalf])     = bw0;
    *(v8bf*)(&wb[buf][srow * 32 + shalf + 8]) = bw1;
  };

  stage(0, 0);
  __syncthreads();

  const int NIT = D_DIM / 32;
  for (int i = 0; i < NIT; ++i) {
    const int cur = i & 1;
    if (i + 1 < NIT) stage((i + 1) * 32, cur ^ 1);      // overlap with WMMAs below

    v16bf bfrag[2];
#pragma unroll
    for (int ns = 0; ns < 2; ++ns)
      bfrag[ns] = load_b(&wb[cur][(wtn * 32 + ns * 16) * 32], 32, lane);
#pragma unroll
    for (int ms = 0; ms < 4; ++ms) {
      v16bf afrag = load_a(&xa[cur][(wtm * 64 + ms * 16) * 32], 32, lane);
#pragma unroll
      for (int ns = 0; ns < 2; ++ns)
        acc[ms][ns] = wmma_bf16(afrag, bfrag[ns], acc[ms][ns]);
    }
    __syncthreads();
  }

  // epilogue: +bias, convert, store bf16
  const int col = lane & 15, rowoff = (lane >> 4) << 3;
  float bv[2];
#pragma unroll
  for (int ns = 0; ns < 2; ++ns) bv[ns] = bias[BN + wtn * 32 + ns * 16 + col];
#pragma unroll
  for (int ms = 0; ms < 4; ++ms)
#pragma unroll
    for (int ns = 0; ns < 2; ++ns)
#pragma unroll
      for (int r = 0; r < 8; ++r) {
        const int rg = BM + wtm * 64 + ms * 16 + rowoff + r;
        const int cg = BN + wtn * 32 + ns * 16 + col;
        out[(size_t)rg * D_DIM + cg] = (__bf16)(acc[ms][ns][r] + bv[ns]);
      }
}

// ---------------------------------------------------------------------------
// Kernel 2: causal flash attention per head (head = contiguous [S][DH] block).
// 4 waves/WG, each wave owns 16 query rows; j streamed in 32-column blocks
// with online softmax. Double-buffered K/V tiles; K tile staged with the
// CDNA5 async global->LDS DMA (ASYNCcnt), V transposed through registers.
// Diagonal is masked (cols >= rows -> -1e9 after scaling) per the reference.
// ---------------------------------------------------------------------------
__global__ __launch_bounds__(128)
void flash_attn(const __bf16* __restrict__ Qb, const __bf16* __restrict__ Kb,
                const __bf16* __restrict__ Vb, float* __restrict__ out) {
  __shared__ __align__(16) __bf16 kt[2][32 * 64];        // K tile  [j][d]
  __shared__ __align__(16) __bf16 vt[2][64 * 32];        // V tile transposed [d][j]
  __shared__ __align__(16) __bf16 pbuf[4 * 16 * 32];     // per-wave P tiles

  const int tid  = threadIdx.x;
  const int lane = tid & 31, wave = tid >> 5;
  const int col = lane & 15, rowoff = (lane >> 4) << 3;
  const size_t hb = (size_t)blockIdx.y * (S_LEN * DH_DIM);
  const int wgi = blockIdx.x * 64;       // workgroup i-base
  const int wit = wgi + wave * 16;       // this wave's i-tile base

  // Q fragments for 16 rows, loaded once, reused over the whole j loop.
  v16bf qa[2];
#pragma unroll
  for (int c = 0; c < 2; ++c)
    qa[c] = load_a(Qb + hb + (size_t)wit * DH_DIM + c * 32, DH_DIM, lane);

  v8f acc[4];
  float mrow[8], lrow[8];
#pragma unroll
  for (int t = 0; t < 4; ++t) acc[t] = vzero8();
#pragma unroll
  for (int r = 0; r < 8; ++r) { mrow[r] = -3.0e38f; lrow[r] = 0.f; }

  __bf16* pb = pbuf + wave * (16 * 32);
  const int sj = tid >> 2, sc = (tid & 3) << 4;          // staging map
  const int njb = (wgi >> 5) + 2;                        // causal j-block count

  auto stageKV = [&](int jblk, int buf) {
    const int jb2 = jblk << 5;
    // K: straight copy via async LDS-DMA (no VGPR round-trip).
    async_copy32(Kb + hb + (size_t)(jb2 + sj) * DH_DIM + sc, &kt[buf][sj * 64 + sc]);
    // V: transpose through registers into [d][j].
    const __bf16* vs = Vb + hb + (size_t)(jb2 + sj) * DH_DIM + sc;
    v8bf v0 = *(const v8bf*)vs, v1 = *(const v8bf*)(vs + 8);
#pragma unroll
    for (int e = 0; e < 8; ++e) {
      vt[buf][(sc + e)     * 32 + sj] = v0[e];
      vt[buf][(sc + e + 8) * 32 + sj] = v1[e];
    }
  };

  stageKV(0, 0);
  wait_async0();
  __syncthreads();

  for (int jb = 0; jb < njb; ++jb) {
    const int jbase = jb << 5;
    const int cur = jb & 1;
    if (jb + 1 < njb) stageKV(jb + 1, cur ^ 1);          // overlap with compute below

    if (jbase < wit + 16) {  // wave-uniform: skip fully-masked blocks (EXEC stays full)
      const __bf16* ktc = kt[cur];
      const __bf16* vtc = vt[cur];
      // scores: two 16x16 j-subtiles, contracting d=64 in two WMMA k-steps
      v8f s0 = wmma_bf16(qa[0], load_b(ktc + 0,            64, lane), vzero8());
      s0     = wmma_bf16(qa[1], load_b(ktc + 32,           64, lane), s0);
      v8f s1 = wmma_bf16(qa[0], load_b(ktc + 16 * 64,      64, lane), vzero8());
      s1     = wmma_bf16(qa[1], load_b(ktc + 16 * 64 + 32, 64, lane), s1);

      float p0[8], p1[8], corr[8];
#pragma unroll
      for (int r = 0; r < 8; ++r) {
        const int irow = wit + rowoff + r;
        float a0 = s0[r] * SM_SCALE; if (jbase + col      >= irow) a0 = NEGV;
        float a1 = s1[r] * SM_SCALE; if (jbase + 16 + col >= irow) a1 = NEGV;
        float rx = fmaxf(a0, a1);
#pragma unroll
        for (int msk = 1; msk < 16; msk <<= 1) rx = fmaxf(rx, __shfl_xor(rx, msk, 32));
        const float mnew = fmaxf(mrow[r], rx);
        corr[r] = __expf(mrow[r] - mnew);
        p0[r] = __expf(a0 - mnew);
        p1[r] = __expf(a1 - mnew);
        float rs = p0[r] + p1[r];
#pragma unroll
        for (int msk = 1; msk < 16; msk <<= 1) rs += __shfl_xor(rs, msk, 32);
        lrow[r] = lrow[r] * corr[r] + rs;
        mrow[r] = mnew;
      }

      // P: C-layout -> A-layout via per-wave LDS (in-order LDS, same wave)
#pragma unroll
      for (int r = 0; r < 8; ++r) {
        pb[(rowoff + r) * 32 + col]      = (__bf16)p0[r];
        pb[(rowoff + r) * 32 + col + 16] = (__bf16)p1[r];
      }
      v16bf pa = load_a(pb, 32, lane);

      // rescale accumulators and accumulate P @ V (B from transposed V tile)
#pragma unroll
      for (int t = 0; t < 4; ++t) {
#pragma unroll
        for (int r = 0; r < 8; ++r) acc[t][r] *= corr[r];
        acc[t] = wmma_bf16(pa, load_b(vtc + t * 16 * 32, 32, lane), acc[t]);
      }
    }

    wait_async0();       // next-buffer K DMA landed in LDS
    __syncthreads();     // next-buffer V stores + everyone done reading cur
  }

  // epilogue: normalize and store f32 (raw-reshape back == same flat layout)
#pragma unroll
  for (int r = 0; r < 8; ++r) {
    const float inv = 1.0f / lrow[r];
    const size_t ob = hb + (size_t)(wit + rowoff + r) * DH_DIM;
#pragma unroll
    for (int t = 0; t < 4; ++t)
      out[ob + t * 16 + col] = acc[t][r] * inv;
  }
}

// ---------------------------------------------------------------------------
extern "C" void kernel_launch(void* const* d_in, const int* in_sizes, int n_in,
                              void* d_out, int out_size, void* d_ws, size_t ws_size,
                              hipStream_t stream) {
  (void)in_sizes; (void)n_in; (void)out_size; (void)ws_size;
  const float* Xq = (const float*)d_in[0];
  const float* Xk = (const float*)d_in[1];
  const float* Xv = (const float*)d_in[2];
  const float* Wq = (const float*)d_in[3];
  const float* bq = (const float*)d_in[4];
  const float* Wk = (const float*)d_in[5];
  const float* bk = (const float*)d_in[6];
  const float* Wv = (const float*)d_in[7];
  const float* bv = (const float*)d_in[8];

  // Workspace: bf16 Q,K,V projections, 3 * 4096*1024 * 2B = 24 MB.
  __bf16* Qb = (__bf16*)d_ws;
  __bf16* Kb = Qb + (size_t)MROWS * D_DIM;
  __bf16* Vb = Kb + (size_t)MROWS * D_DIM;

  dim3 pgrid(D_DIM / 128, MROWS / 128);  // (8, 32)
  proj_gemm<<<pgrid, 256, 0, stream>>>(Xq, Wq, bq, Qb);
  proj_gemm<<<pgrid, 256, 0, stream>>>(Xk, Wk, bk, Kb);
  proj_gemm<<<pgrid, 256, 0, stream>>>(Xv, Wv, bv, Vb);

  dim3 agrid(S_LEN / 64, BH);            // (32, 32)
  flash_attn<<<agrid, 128, 0, stream>>>(Qb, Kb, Vb, (float*)d_out);
}
